// GraphConv_23596550324575
// MI455X (gfx1250) — matrix operand order repaired and checked by reading, not compiled, not run on hardware
//
#include <hip/hip_runtime.h>

// ---------------- problem constants (from reference) ----------------
#define N_ENT   8192
#define N_USR   16384
#define CH      64
#define N_EDGE  1000000
#define NNZ     500000
#define N_RELW  8          // N_REL - 1 rows in weight
#define TOPK    10
#define LAMBDA  0.5f
#define CHUNK   256        // sim columns staged in LDS per iteration

typedef __bf16 bf16_t;
typedef bf16_t v16bf __attribute__((ext_vector_type(16)));
typedef bf16_t v8bf  __attribute__((ext_vector_type(8)));
typedef float  v8f   __attribute__((ext_vector_type(8)));

__device__ __forceinline__ float wave_sum(float v) {
#pragma unroll
  for (int o = 16; o > 0; o >>= 1) v += __shfl_xor(v, o, 32);
  return v;
}

// ---------------- generic fill / copy (float4 granularity) ----------------
__global__ void fill4_kernel(float4* __restrict__ p, float v, int n4) {
  int i = blockIdx.x * blockDim.x + threadIdx.x;
  if (i < n4) p[i] = make_float4(v, v, v, v);
}
__global__ void copy4_kernel(float4* __restrict__ dst, const float4* __restrict__ src, int n4) {
  int i = blockIdx.x * blockDim.x + threadIdx.x;
  if (i < n4) dst[i] = src[i];
}

// ---------------- row-normalize -> bf16 (one wave32 per 64-ch row) ----------------
__global__ __launch_bounds__(256) void rownorm_bf16_kernel(const float* __restrict__ x,
                                                           bf16_t* __restrict__ out, int nrows) {
  const int lane = threadIdx.x & 31;
  const int row  = blockIdx.x * 8 + (threadIdx.x >> 5);
  if (row >= nrows) return;
  const float2 v = *(const float2*)(x + row * CH + lane * 2);
  float ss = wave_sum(v.x * v.x + v.y * v.y);
  const float inv = rsqrtf(fmaxf(ss, 1e-24f));
  out[row * CH + lane * 2]     = (bf16_t)(v.x * inv);
  out[row * CH + lane * 2 + 1] = (bf16_t)(v.y * inv);
}

// ---------------- l2norm + residual accumulate (one wave per row) ----------------
__global__ __launch_bounds__(256) void finalize_kernel(const float* __restrict__ agg,
                                                       float* __restrict__ nxt,
                                                       float* __restrict__ res, int nrows) {
  const int lane = threadIdx.x & 31;
  const int row  = blockIdx.x * 8 + (threadIdx.x >> 5);
  if (row >= nrows) return;
  const float2 v = *(const float2*)(agg + row * CH + lane * 2);
  float ss = wave_sum(v.x * v.x + v.y * v.y);
  const float scale = 1.0f / fmaxf(sqrtf(ss), 1e-12f);
  const float n0 = v.x * scale, n1 = v.y * scale;
  nxt[row * CH + lane * 2]     = n0;
  nxt[row * CH + lane * 2 + 1] = n1;
  res[row * CH + lane * 2]     += n0;
  res[row * CH + lane * 2 + 1] += n1;
}

// ---------------- WMMA sim + wave-parallel running top-k ----------------
__global__ __launch_bounds__(128) void sim_topk_kernel(const bf16_t* __restrict__ cn,
                                                       float* __restrict__ kval,
                                                       int* __restrict__ kind,
                                                       float* __restrict__ rowsum) {
  __shared__ float s_chunk[16][CHUNK + 1];
  __shared__ float s_topv[16][TOPK];
  __shared__ int   s_topi[16][TOPK];

  const int tid  = threadIdx.x;
  const int lane = tid & 31;
  const int wave = tid >> 5;
  const int lm   = lane & 15;
  const bool hi  = lane >= 16;
  const int rb   = blockIdx.x;                 // 16-row block

  if (tid < 16) {
#pragma unroll
    for (int q = 0; q < TOPK; ++q) { s_topv[tid][q] = -1e30f; s_topi[tid][q] = 0; }
  }
  float minv[4];                               // running per-row min (wave-uniform)
#pragma unroll
  for (int rr = 0; rr < 4; ++rr) minv[rr] = -1e30f;
  __syncthreads();

  // A fragments (16x32 bf16, contiguous K-halves), reused for every column tile
  const bf16_t* arow = cn + (rb * 16 + lm) * CH;
  v16bf afrag[2];
#pragma unroll
  for (int kt = 0; kt < 2; ++kt) {
    const int sel = hi ? 16 : 0;
    v8bf lo = *(const v8bf*)(arow + kt * 32 + sel);
    v8bf hh = *(const v8bf*)(arow + kt * 32 + 8 + sel);
    afrag[kt] = __builtin_shufflevector(lo, hh, 0,1,2,3,4,5,6,7,8,9,10,11,12,13,14,15);
  }
  const int koff = hi ? 16 : 0;

  for (int c = 0; c < (N_ENT / CHUNK); ++c) {
    // ---- WMMA phase: batch-load all 8 B fragments, then 8 interleaved WMMAs ----
    v16bf bfrag[4][2];
#pragma unroll
    for (int t = 0; t < 4; ++t) {
      const int tloc = wave * 4 + t;
      const bf16_t* brow = cn + (c * CHUNK + tloc * 16 + lm) * CH;
      bfrag[t][0] = *(const v16bf*)(brow + koff);
      bfrag[t][1] = *(const v16bf*)(brow + 32 + koff);
    }
    v8f acc[4];
#pragma unroll
    for (int t = 0; t < 4; ++t) acc[t] = (v8f){};
#pragma unroll
    for (int kt = 0; kt < 2; ++kt) {
#pragma unroll
      for (int t = 0; t < 4; ++t) {            // distance-4 dep chain per acc[t]
        acc[t] = __builtin_amdgcn_wmma_f32_16x16x32_bf16(false, afrag[kt], false, bfrag[t][kt],
                                                         (short)0, acc[t], false, false);
      }
    }
#pragma unroll
    for (int t = 0; t < 4; ++t) {
      const int cl = (wave * 4 + t) * 16 + lm;
#pragma unroll
      for (int r = 0; r < 8; ++r) s_chunk[hi ? (r + 8) : r][cl] = acc[t][r];
    }
    __syncthreads();

    // ---- top-k phase: wave w owns rows 4w..4w+3; ballot-filtered insertion ----
    {
      const int base = c * CHUNK;
#pragma unroll
      for (int rr = 0; rr < 4; ++rr) {
        const int row = wave * 4 + rr;
        float mv = minv[rr];
        for (int s = 0; s < CHUNK / 32; ++s) {
          const float v = s_chunk[row][s * 32 + lane];
          unsigned long long mask = __ballot(v > mv);
          while (mask) {
            const int src = __ffsll(mask) - 1;
            mask &= mask - 1;
            const float cv = __shfl(v, src, 32);
            float nm = mv;
            if (lane == 0) {
              float m = s_topv[row][0]; int mp = 0;
#pragma unroll
              for (int q = 1; q < TOPK; ++q) {
                const float tv = s_topv[row][q];
                if (tv < m) { m = tv; mp = q; }
              }
              if (cv > m) {                    // re-check vs fresh min (mask may be stale)
                s_topv[row][mp] = cv;
                s_topi[row][mp] = base + s * 32 + src;
                m = s_topv[row][0];
#pragma unroll
                for (int q = 1; q < TOPK; ++q) m = fminf(m, s_topv[row][q]);
              }
              nm = m;
            }
            mv = __shfl(nm, 0, 32);
          }
        }
        minv[rr] = mv;
      }
    }
    __syncthreads();
  }

  if (tid < 16) {
    const int row = rb * 16 + tid;
    float s = 0.f;
#pragma unroll
    for (int q = 0; q < TOPK; ++q) {
      kval[row * TOPK + q] = s_topv[tid][q];
      kind[row * TOPK + q] = s_topi[tid][q];
      s += s_topv[tid][q];
    }
    rowsum[row] = s;
  }
}

// ---------------- sparse scatter into dense adj: D^-1/2 A D^-1/2 ----------------
__global__ void adj_scatter_kernel(const float* __restrict__ kval, const int* __restrict__ kind,
                                   const float* __restrict__ rowsum, float coeff,
                                   float* __restrict__ adj) {
  int idx = blockIdx.x * blockDim.x + threadIdx.x;
  if (idx >= N_ENT * TOPK) return;
  const int row = idx / TOPK;
  const int col = kind[idx];
  const float v = kval[idx];
  const float dr = rsqrtf(fmaxf(rowsum[row], 1e-30f));
  const float dc = rsqrtf(fmaxf(rowsum[col], 1e-30f));
  atomicAdd(adj + (size_t)row * N_ENT + col, coeff * dr * v * dc);
}

// ---------------- edge pass A: att = |h*r|^2 * |t*r|^2, segment max ----------------
__global__ void edge_att_kernel(const int* __restrict__ head, const int* __restrict__ tail,
                                const int* __restrict__ etype, const float* __restrict__ ent,
                                const float* __restrict__ W, float* __restrict__ att,
                                float* __restrict__ m, int ne) {
  int e = blockIdx.x * blockDim.x + threadIdx.x;
  if (e >= ne) return;
  const int h = head[e], t = tail[e], r = etype[e] - 1;
  const float4* he = (const float4*)(ent + h * CH);
  const float4* te = (const float4*)(ent + t * CH);
  const float4* wr = (const float4*)(W + r * CH);
  float sa = 0.f, sb = 0.f;
#pragma unroll
  for (int i = 0; i < 16; ++i) {
    const float4 w = wr[i], a = he[i], b = te[i];
    float x;
    x = a.x * w.x; sa += x * x;  x = a.y * w.y; sa += x * x;
    x = a.z * w.z; sa += x * x;  x = a.w * w.w; sa += x * x;
    x = b.x * w.x; sb += x * x;  x = b.y * w.y; sb += x * x;
    x = b.z * w.z; sb += x * x;  x = b.w * w.w; sb += x * x;
  }
  const float a = sa * sb;                       // (|h r| * |t r|)^2, >= 0
  att[e] = a;
  atomicMax((unsigned int*)(m + h), __float_as_uint(a));  // valid for non-negative floats
}

// ---------------- edge pass B: e = exp(att - m[head]); denom += e ----------------
__global__ void edge_exp_kernel(const int* __restrict__ head, float* __restrict__ att,
                                const float* __restrict__ m, float* __restrict__ denom, int ne) {
  int e = blockIdx.x * blockDim.x + threadIdx.x;
  if (e >= ne) return;
  const int h = head[e];
  const float ex = __expf(att[e] - m[h]);
  att[e] = ex;
  atomicAdd(denom + h, ex);
}

// ---------------- edge pass C: entity_agg[head] += (e/denom) * t_emb * rel ----------------
__global__ void edge_scatter_kernel(const int* __restrict__ head, const int* __restrict__ tail,
                                    const int* __restrict__ etype, const float* __restrict__ ent,
                                    const float* __restrict__ W, const float* __restrict__ att,
                                    const float* __restrict__ denom, float* __restrict__ eagg, int ne) {
  int idx = blockIdx.x * blockDim.x + threadIdx.x;
  const int e = idx >> 4, cg = idx & 15;
  if (e >= ne) return;
  const int h = head[e], t = tail[e], r = etype[e] - 1;
  const float w = att[e] / denom[h];
  const float4 tv = *(const float4*)(ent + t * CH + cg * 4);
  const float4 wv = *(const float4*)(W + r * CH + cg * 4);
  float* dst = eagg + h * CH + cg * 4;
  atomicAdd(dst + 0, w * tv.x * wv.x);
  atomicAdd(dst + 1, w * tv.y * wv.y);
  atomicAdd(dst + 2, w * tv.z * wv.z);
  atomicAdd(dst + 3, w * tv.w * wv.w);
}

// ---------------- user_agg = interact_mat @ entity_emb (scatter SpMM) ----------------
__global__ void spmm_kernel(const int* __restrict__ irows, const int* __restrict__ icols,
                            const float* __restrict__ ivals, const float* __restrict__ ent,
                            float* __restrict__ uagg, int nnz) {
  int idx = blockIdx.x * blockDim.x + threadIdx.x;
  const int n = idx >> 4, cg = idx & 15;
  if (n >= nnz) return;
  const int rr = irows[n], cc = icols[n];
  const float v = ivals[n];
  const float4 ev = *(const float4*)(ent + cc * CH + cg * 4);
  float* dst = uagg + rr * CH + cg * 4;
  atomicAdd(dst + 0, v * ev.x);
  atomicAdd(dst + 1, v * ev.y);
  atomicAdd(dst + 2, v * ev.z);
  atomicAdd(dst + 3, v * ev.w);
}

// ---------------- user gate: uagg *= 1 + softmax(u @ W^T) @ W (wave per user) ----------------
__global__ __launch_bounds__(256) void user_gate_kernel(const float* __restrict__ user,
                                                        const float* __restrict__ W,
                                                        float* __restrict__ uagg, int nusr) {
  const int lane = threadIdx.x & 31;
  const int u = blockIdx.x * 8 + (threadIdx.x >> 5);
  if (u >= nusr) return;
  const float u0 = user[u * CH + lane];
  const float u1 = user[u * CH + 32 + lane];
  float d[N_RELW];
#pragma unroll
  for (int r = 0; r < N_RELW; ++r)
    d[r] = wave_sum(u0 * W[r * CH + lane] + u1 * W[r * CH + 32 + lane]);
  float mx = d[0];
#pragma unroll
  for (int r = 1; r < N_RELW; ++r) mx = fmaxf(mx, d[r]);
  float s = 0.f;
#pragma unroll
  for (int r = 0; r < N_RELW; ++r) { d[r] = __expf(d[r] - mx); s += d[r]; }
  const float invs = 1.0f / s;
  float p0 = 0.f, p1 = 0.f;
#pragma unroll
  for (int r = 0; r < N_RELW; ++r) {
    const float sc = d[r] * invs;
    p0 += sc * W[r * CH + lane];
    p1 += sc * W[r * CH + 32 + lane];
  }
  uagg[u * CH + lane]      *= (1.0f + p0);
  uagg[u * CH + 32 + lane] *= (1.0f + p1);
}

// =======================================================================
extern "C" void kernel_launch(void* const* d_in, const int* in_sizes, int n_in,
                              void* d_out, int out_size, void* d_ws, size_t ws_size,
                              hipStream_t stream) {
  (void)in_sizes; (void)n_in; (void)out_size; (void)ws_size;
  const float* user_emb = (const float*)d_in[0];
  const float* ent_emb  = (const float*)d_in[1];
  const int*   head     = (const int*)d_in[2];
  const int*   tail     = head + N_EDGE;
  const int*   etype    = (const int*)d_in[3];
  const int*   irows    = (const int*)d_in[4];
  const int*   icols    = irows + NNZ;
  const float* ivals    = (const float*)d_in[5];
  const float* weight   = (const float*)d_in[6];

  float* out_ent = (float*)d_out;                            // 8192*64
  float* out_usr = out_ent + N_ENT * CH;                     // 16384*64
  float* adj     = out_usr + N_USR * CH;                     // 8192*8192

  // small persistent scratch in d_ws (~2.4 MB)
  char* wsb = (char*)d_ws;
  bf16_t* cnb  = (bf16_t*)wsb;                               // 8192*64 bf16
  float* kval0 = (float*)(wsb + (size_t)N_ENT * CH * 2);
  int*   kind0 = (int*)(kval0 + N_ENT * TOPK);
  float* rs0   = (float*)(kind0 + N_ENT * TOPK);
  float* kval1 = rs0 + N_ENT;
  int*   kind1 = (int*)(kval1 + N_ENT * TOPK);
  float* rs1   = (float*)(kind1 + N_ENT * TOPK);

  // big hop scratch overlaid on the (still unused) item_adj output region
  float* ent_a = adj;
  float* ent_b = ent_a + N_ENT * CH;
  float* usr_a = ent_b + N_ENT * CH;
  float* usr_b = usr_a + N_USR * CH;
  float* eagg  = usr_b + N_USR * CH;
  float* uagg  = eagg + N_ENT * CH;
  float* attb  = uagg + N_USR * CH;          // 1M floats (padded slot 1<<20)
  float* mbuf  = attb + (1 << 20);
  float* dnm   = mbuf + N_ENT;

  const int TB = 256;
  const int ENT_F  = N_ENT * CH, USR_F = N_USR * CH;

  // residual init + hop-0 inputs
  copy4_kernel<<<(ENT_F / 4 + TB - 1) / TB, TB, 0, stream>>>((float4*)out_ent, (const float4*)ent_emb,  ENT_F / 4);
  copy4_kernel<<<(USR_F / 4 + TB - 1) / TB, TB, 0, stream>>>((float4*)out_usr, (const float4*)user_emb, USR_F / 4);
  copy4_kernel<<<(ENT_F / 4 + TB - 1) / TB, TB, 0, stream>>>((float4*)ent_a,   (const float4*)ent_emb,  ENT_F / 4);
  copy4_kernel<<<(USR_F / 4 + TB - 1) / TB, TB, 0, stream>>>((float4*)usr_a,   (const float4*)user_emb, USR_F / 4);

  float *ecur = ent_a, *enxt = ent_b, *ucur = usr_a, *unxt = usr_b;
  for (int hop = 0; hop < 2; ++hop) {
    fill4_kernel<<<(N_ENT / 4 + TB - 1) / TB, TB, 0, stream>>>((float4*)mbuf, 0.f, N_ENT / 4);
    fill4_kernel<<<(N_ENT / 4 + TB - 1) / TB, TB, 0, stream>>>((float4*)dnm,  0.f, N_ENT / 4);
    fill4_kernel<<<(ENT_F / 4 + TB - 1) / TB, TB, 0, stream>>>((float4*)eagg, 0.f, ENT_F / 4);
    fill4_kernel<<<(USR_F / 4 + TB - 1) / TB, TB, 0, stream>>>((float4*)uagg, 0.f, USR_F / 4);

    edge_att_kernel<<<(N_EDGE + TB - 1) / TB, TB, 0, stream>>>(head, tail, etype, ecur, weight, attb, mbuf, N_EDGE);
    edge_exp_kernel<<<(N_EDGE + TB - 1) / TB, TB, 0, stream>>>(head, attb, mbuf, dnm, N_EDGE);
    edge_scatter_kernel<<<((size_t)N_EDGE * 16 + TB - 1) / TB, TB, 0, stream>>>(head, tail, etype, ecur, weight,
                                                                                attb, dnm, eagg, N_EDGE);
    spmm_kernel<<<((size_t)NNZ * 16 + TB - 1) / TB, TB, 0, stream>>>(irows, icols, ivals, ecur, uagg, NNZ);
    user_gate_kernel<<<N_USR / 8, TB, 0, stream>>>(ucur, weight, uagg, N_USR);

    finalize_kernel<<<N_ENT / 8, TB, 0, stream>>>(eagg, enxt, out_ent, N_ENT);
    finalize_kernel<<<N_USR / 8, TB, 0, stream>>>(uagg, unxt, out_usr, N_USR);

    float* tmp;
    tmp = ecur; ecur = enxt; enxt = tmp;
    tmp = ucur; ucur = unxt; unxt = tmp;
  }

  // build_adj(entity_emb)  and  build_adj(entity_res_emb): WMMA sim + top-k
  rownorm_bf16_kernel<<<N_ENT / 8, TB, 0, stream>>>(ent_emb, cnb, N_ENT);
  sim_topk_kernel<<<N_ENT / 16, 128, 0, stream>>>(cnb, kval0, kind0, rs0);
  rownorm_bf16_kernel<<<N_ENT / 8, TB, 0, stream>>>(out_ent, cnb, N_ENT);
  sim_topk_kernel<<<N_ENT / 16, 128, 0, stream>>>(cnb, kval1, kind1, rs1);

  // item_adj = (1-L)*adj(res) + L*adj(origin): zero 256MB then sparse scatter
  const size_t adjN4 = (size_t)N_ENT * N_ENT / 4;
  fill4_kernel<<<(adjN4 + TB - 1) / TB, TB, 0, stream>>>((float4*)adj, 0.f, (int)adjN4);
  adj_scatter_kernel<<<(N_ENT * TOPK + TB - 1) / TB, TB, 0, stream>>>(kval0, kind0, rs0, LAMBDA, adj);
  adj_scatter_kernel<<<(N_ENT * TOPK + TB - 1) / TB, TB, 0, stream>>>(kval1, kind1, rs1, 1.0f - LAMBDA, adj);
}
